// Gemma2Attention_32925219291583
// MI455X (gfx1250) — compile-verified, matrix operand order checked by reading, not run on hardware
//
#include <hip/hip_runtime.h>
#include <cstdint>
#include <cstddef>

// ---------------------------------------------------------------------------
// Gemma2 attention layer for gfx1250 (MI455X), bf16 WMMA pipeline with
// async global->LDS double buffering (GLOBAL_LOAD_ASYNC_TO_LDS_B128).
// B=1, S=4096, HID=2048, NH=8, NKV=4, HD=256, softcap=50, causal (window==S).
// ---------------------------------------------------------------------------

typedef __bf16 bf16;
typedef __attribute__((ext_vector_type(16))) __bf16 v16bf;
typedef __attribute__((ext_vector_type(8)))  __bf16 v8bf;
typedef __attribute__((ext_vector_type(8)))  float  v8f;

static constexpr int S    = 4096;
static constexpr int HID  = 2048;
static constexpr int NH   = 8;
static constexpr int NKV  = 4;
static constexpr int HD   = 256;
static constexpr float SCALING  = 0.0625f;       // 256^-0.5
static constexpr float SOFT_CAP = 50.0f;

// ---- CDNA5 async global->LDS copy (16B per lane), tracked by ASYNCcnt ------
// LDS destination address = low 32 bits of the generic shared pointer
// (shared aperture keeps the LDS byte offset in addr[31:0], ISA 10.2).
__device__ __forceinline__ void async_copy16(bf16* lds_dst, const bf16* gsrc) {
  asm volatile("global_load_async_to_lds_b128 %0, %1, off"
               :: "v"((uint32_t)(uintptr_t)lds_dst), "v"(gsrc)
               : "memory");
}
__device__ __forceinline__ void wait_async0() {
  asm volatile("s_wait_asynccnt 0x0" ::: "memory");
}

// A-fragment / Bt-fragment loader for v_wmma_f32_16x16x32_bf16.
// lane L holds row (L%16); 16 halves split in two 16-byte chunks at
// k-offsets base and base+16, base = (L/16)*8.  (ISA 7.12.2, 16-bit A 16x32)
__device__ __forceinline__ v16bf load_frag(const bf16* base, int ld) {
  const int lane = threadIdx.x & 31;
  const bf16* p = base + (size_t)(lane & 15) * ld + ((lane >> 4) << 3);
  v8bf lo = *(const v8bf*)(p);
  v8bf hi = *(const v8bf*)(p + 16);
  v16bf r;
#pragma unroll
  for (int i = 0; i < 8; ++i) { r[i] = lo[i]; r[i + 8] = hi[i]; }
  return r;
}

__device__ __forceinline__ v8f wmma_bf16(v16bf a, v16bf b, v8f c) {
  return __builtin_amdgcn_wmma_f32_16x16x32_bf16(false, a, false, b,
                                                 (short)0, c, false, false);
}

// --------------------------- conversion kernels -----------------------------

__global__ __launch_bounds__(256) void cvt_f32_bf16(const float* __restrict__ src,
                                                    bf16* __restrict__ dst, int n) {
  int i = blockIdx.x * blockDim.x + threadIdx.x;
  if (i < n) dst[i] = (bf16)src[i];
}

// src: row-major [K][N] fp32 -> dst: row-major [N][K] bf16 (transposed)
__global__ __launch_bounds__(256) void cvt_f32_bf16_T(const float* __restrict__ src,
                                                      bf16* __restrict__ dst,
                                                      int K, int N) {
  int i = blockIdx.x * blockDim.x + threadIdx.x;
  if (i < K * N) {
    int k = i / N, n = i - k * N;
    dst[(size_t)n * K + k] = (bf16)src[i];
  }
}

// ------------------------------- RoPE ---------------------------------------

__global__ __launch_bounds__(256) void rope_kernel(bf16* __restrict__ buf,
                                                   const long long* __restrict__ pos,
                                                   int nh) {
  int i = blockIdx.x * blockDim.x + threadIdx.x;   // S * nh * 128 threads
  int per_s = nh * 128;
  if (i >= S * per_s) return;
  int s = i / per_s;
  int r = i - s * per_s;
  int h = r >> 7;
  int d = r & 127;
  size_t base = (size_t)s * (nh * HD) + (size_t)h * HD + d;
  float p = (float)pos[s];
  // inv_freq = 10000^(-d/128)
  float freq = p * __expf(-(float)d * (9.210340371976184f / 128.0f));
  float cs = __cosf(freq), sn = __sinf(freq);
  float x1 = (float)buf[base];
  float x2 = (float)buf[base + 128];
  buf[base]       = (bf16)(x1 * cs - x2 * sn);
  buf[base + 128] = (bf16)(x2 * cs + x1 * sn);
}

// --------------------------- bf16 WMMA GEMM ---------------------------------
// C[M][N] = A[M][K] * B[K][N], with B supplied transposed: Bt[N][K].
// OUT_MODE: 0 = bf16 row-major, 1 = bf16 transposed (C[n*ldc+m]), 2 = f32.
// 128 threads = 4 waves; WG tile 64x64; each wave 32x32 (2x2 WMMA accums).
// K staged in 32-deep slabs via async-to-LDS, double buffered.

template <int OUT_MODE>
__global__ __launch_bounds__(128) void gemm_wmma(const bf16* __restrict__ A,
                                                 const bf16* __restrict__ Bt,
                                                 void* __restrict__ Cout,
                                                 int M, int N, int K, int ldc) {
  __shared__ __align__(16) bf16 As[2][64 * 32];
  __shared__ __align__(16) bf16 Bs[2][64 * 32];
  const int m0 = blockIdx.y * 64;
  const int n0 = blockIdx.x * 64;
  const int t = threadIdx.x;
  const int w = t >> 5, wy = w >> 1, wx = w & 1;

  v8f acc[2][2];
  const v8f vzero = {0.f, 0.f, 0.f, 0.f, 0.f, 0.f, 0.f, 0.f};
#pragma unroll
  for (int i = 0; i < 2; ++i)
#pragma unroll
    for (int j = 0; j < 2; ++j) acc[i][j] = vzero;

  auto stage = [&](int buf, int k0) {
#pragma unroll
    for (int j = 0; j < 2; ++j) {
      int c = t * 2 + j;                       // 0..255 : 64 rows x 4 chunks
      int row = c >> 2, col = (c & 3) << 3;
      async_copy16(&As[buf][row * 32 + col],
                   &A[(size_t)(m0 + row) * K + k0 + col]);
      async_copy16(&Bs[buf][row * 32 + col],
                   &Bt[(size_t)(n0 + row) * K + k0 + col]);
    }
  };

  const int nslab = K >> 5;
  stage(0, 0);
  for (int s = 0; s < nslab; ++s) {
    wait_async0();
    __syncthreads();                            // slab s fully resident
    if (s + 1 < nslab) stage((s + 1) & 1, (s + 1) << 5);

    const bf16* as = As[s & 1];
    const bf16* bs = Bs[s & 1];
    v16bf af0 = load_frag(&as[(wy * 32 + 0) * 32], 32);
    v16bf af1 = load_frag(&as[(wy * 32 + 16) * 32], 32);
    v16bf bf0 = load_frag(&bs[(wx * 32 + 0) * 32], 32);
    v16bf bf1 = load_frag(&bs[(wx * 32 + 16) * 32], 32);

    acc[0][0] = wmma_bf16(af0, bf0, acc[0][0]);
    acc[0][1] = wmma_bf16(af0, bf1, acc[0][1]);
    acc[1][0] = wmma_bf16(af1, bf0, acc[1][0]);
    acc[1][1] = wmma_bf16(af1, bf1, acc[1][1]);
    // no trailing barrier: next iteration's wait+barrier protects the buffers
  }

  // epilogue: C 16x16 layout -> lane L: col = L%16, rows r + 8*(L/16)
  const int lane = t & 31;
  const int half = lane >> 4, nc = lane & 15;
#pragma unroll
  for (int i = 0; i < 2; ++i)
#pragma unroll
    for (int j = 0; j < 2; ++j)
#pragma unroll
      for (int r = 0; r < 8; ++r) {
        int mrow = m0 + wy * 32 + i * 16 + r + half * 8;
        int ncol = n0 + wx * 32 + j * 16 + nc;
        float v = acc[i][j][r];
        if (OUT_MODE == 0)
          ((bf16*)Cout)[(size_t)mrow * ldc + ncol] = (bf16)v;
        else if (OUT_MODE == 1)
          ((bf16*)Cout)[(size_t)ncol * ldc + mrow] = (bf16)v;
        else
          ((float*)Cout)[(size_t)mrow * ldc + ncol] = v;
      }
}

// ------------------------- flash attention (bf16) ---------------------------
// Q  : [S][NH*HD]  bf16 (RoPE applied)
// Kb : [S][NKV*HD] bf16 (RoPE applied)
// Vt : [NKV*HD][S] bf16 (transposed V projection)
// O  : [S][NH*HD]  bf16
// Grid: (S/64, NH), 128 threads. Each wave: 16 queries x full head dim 256.
// K/V tiles (32 keys) streamed via async-to-LDS, double buffered.

__global__ __launch_bounds__(128) void attn_wmma(const bf16* __restrict__ Q,
                                                 const bf16* __restrict__ Kb,
                                                 const bf16* __restrict__ Vt,
                                                 bf16* __restrict__ O) {
  __shared__ __align__(16) bf16 Ks[2][32 * 256];   // 32 keys x 256 dims
  __shared__ __align__(16) bf16 Vs[2][256 * 32];   // 256 dims x 32 keys
  __shared__ __align__(16) bf16 Ps[4][16 * 32];    // per-wave P tile

  const int h = blockIdx.y;
  const int kvh = h >> 1;                          // n_rep = 2
  const int q0 = blockIdx.x * 64;
  const int t = threadIdx.x, w = t >> 5, lane = t & 31;
  const int half = lane >> 4, nidx = lane & 15;
  const int qb = q0 + w * 16;

  // preload Q fragments: 8 chunks of 16x32 over head dim 256
  v16bf qf[8];
#pragma unroll
  for (int c = 0; c < 8; ++c)
    qf[c] = load_frag(&Q[(size_t)qb * (NH * HD) + h * HD + c * 32], NH * HD);

  const v8f vzero = {0.f, 0.f, 0.f, 0.f, 0.f, 0.f, 0.f, 0.f};
  v8f o[16];
#pragma unroll
  for (int i = 0; i < 16; ++i) o[i] = vzero;
  float mrow[8], lrow[8];
#pragma unroll
  for (int r = 0; r < 8; ++r) { mrow[r] = -INFINITY; lrow[r] = 0.f; }

  auto stage = [&](int buf, int kb) {
#pragma unroll
    for (int j = 0; j < 4; ++j) {                  // K: 512 chunks (32 x 16)
      int c = t * 4 + j;
      int row = c >> 4, col = (c & 15) << 3;
      async_copy16(&Ks[buf][row * 256 + col],
                   &Kb[(size_t)(kb + row) * (NKV * HD) + kvh * HD + col]);
    }
#pragma unroll
    for (int j = 0; j < 8; ++j) {                  // Vt: 1024 chunks (256 x 4)
      int c = t * 8 + j;
      int row = c >> 2, col = (c & 3) << 3;
      async_copy16(&Vs[buf][row * 32 + col],
                   &Vt[(size_t)(kvh * HD + row) * S + kb + col]);
    }
  };

  const int ntiles = (q0 + 64) >> 5;               // causal bound for the WG
  stage(0, 0);
  for (int kt = 0; kt < ntiles; ++kt) {
    const int kb = kt * 32;
    const int buf = kt & 1;
    wait_async0();
    __syncthreads();                               // tile kt resident
    if (kt + 1 < ntiles) stage(buf ^ 1, kb + 32);

    const bf16* ks = Ks[buf];
    const bf16* vs = Vs[buf];

    // scores: S0 = Q*K^T (keys 0..15), S1 = keys 16..31; pipelined frag loads
    v8f s0 = vzero, s1 = vzero;
    v16bf k0 = load_frag(&ks[0 * 256 + 0], 256);
    v16bf k1 = load_frag(&ks[16 * 256 + 0], 256);
#pragma unroll
    for (int c = 0; c < 8; ++c) {
      v16bf nk0 = k0, nk1 = k1;
      if (c < 7) {
        nk0 = load_frag(&ks[0 * 256 + (c + 1) * 32], 256);
        nk1 = load_frag(&ks[16 * 256 + (c + 1) * 32], 256);
      }
      s0 = wmma_bf16(qf[c], k0, s0);
      s1 = wmma_bf16(qf[c], k1, s1);
      k0 = nk0; k1 = nk1;
    }

    // softcap + causal mask + online softmax
#pragma unroll
    for (int r = 0; r < 8; ++r) {
      const int qi = qb + r + half * 8;
      float x0 = SOFT_CAP * tanhf(s0[r] * (SCALING / SOFT_CAP));
      float x1 = SOFT_CAP * tanhf(s1[r] * (SCALING / SOFT_CAP));
      if (kb + nidx > qi)       x0 = -1e30f;
      if (kb + 16 + nidx > qi)  x1 = -1e30f;

      float rmax = fmaxf(x0, x1);
      rmax = fmaxf(rmax, __shfl_xor(rmax, 1, 32));
      rmax = fmaxf(rmax, __shfl_xor(rmax, 2, 32));
      rmax = fmaxf(rmax, __shfl_xor(rmax, 4, 32));
      rmax = fmaxf(rmax, __shfl_xor(rmax, 8, 32));

      float mnew = fmaxf(mrow[r], rmax);
      float corr = __expf(mrow[r] - mnew);
      float p0 = __expf(x0 - mnew);
      float p1 = __expf(x1 - mnew);
      float rs = p0 + p1;
      rs += __shfl_xor(rs, 1, 32);
      rs += __shfl_xor(rs, 2, 32);
      rs += __shfl_xor(rs, 4, 32);
      rs += __shfl_xor(rs, 8, 32);

      lrow[r] = lrow[r] * corr + rs;
      mrow[r] = mnew;
#pragma unroll
      for (int ds = 0; ds < 16; ++ds) o[ds][r] *= corr;

      // relayout P (C-layout lane=col) into row-major LDS for A-fragments
      Ps[w][(r + half * 8) * 32 + nidx]      = (bf16)p0;
      Ps[w][(r + half * 8) * 32 + 16 + nidx] = (bf16)p1;
    }

    // O += P (16x32) * V (32x256): 16 WMMAs, pipelined V-fragment loads
    v16bf pf = load_frag(&Ps[w][0], 32);
    v16bf vf = load_frag(&vs[0], 32);
#pragma unroll
    for (int ds = 0; ds < 16; ++ds) {
      v16bf nvf = vf;
      if (ds < 15) nvf = load_frag(&vs[((ds + 1) * 16) * 32], 32);
      o[ds] = wmma_bf16(pf, vf, o[ds]);
      vf = nvf;
    }
    // no trailing barrier: next iteration's wait+barrier protects the buffers
  }

  // epilogue: normalize and store
#pragma unroll
  for (int r = 0; r < 8; ++r) {
    const int qi = qb + r + half * 8;
    const float inv = 1.0f / lrow[r];
#pragma unroll
    for (int ds = 0; ds < 16; ++ds)
      O[(size_t)qi * (NH * HD) + h * HD + ds * 16 + nidx] =
          (bf16)(o[ds][r] * inv);
  }
}

// ------------------------------- launcher -----------------------------------

extern "C" void kernel_launch(void* const* d_in, const int* in_sizes, int n_in,
                              void* d_out, int out_size, void* d_ws, size_t ws_size,
                              hipStream_t stream) {
  const float*      hidden = (const float*)d_in[0];
  // d_in[1] = attention_mask (unused: causal + window==S implemented in-kernel)
  const long long*  posids = (const long long*)d_in[2];
  const float*      wq = (const float*)d_in[3];
  const float*      wk = (const float*)d_in[4];
  const float*      wv = (const float*)d_in[5];
  const float*      wo = (const float*)d_in[6];
  float*            out = (float*)d_out;

  // workspace carve-up (bf16 buffers), 256B aligned
  size_t off = 0;
  auto carve = [&](size_t bytes) -> bf16* {
    bf16* p = (bf16*)((char*)d_ws + off);
    off += (bytes + 255) & ~(size_t)255;
    return p;
  };
  bf16* Hb   = carve((size_t)S * HID * 2);          // hidden bf16
  bf16* WqT  = carve((size_t)HID * (NH * HD) * 2);  // wq^T [2048][2048]
  bf16* WkT  = carve((size_t)HID * (NKV * HD) * 2); // wk^T [1024][2048]
  bf16* WvT  = carve((size_t)HID * (NKV * HD) * 2); // wv^T [1024][2048]
  bf16* WoT  = carve((size_t)(NH * HD) * HID * 2);  // wo^T [2048][2048]
  bf16* Qb   = carve((size_t)S * (NH * HD) * 2);    // [S][2048]
  bf16* Kb   = carve((size_t)S * (NKV * HD) * 2);   // [S][1024]
  bf16* Vt   = carve((size_t)(NKV * HD) * S * 2);   // [1024][S]
  bf16* AttO = carve((size_t)S * (NH * HD) * 2);    // [S][2048]
  (void)ws_size;

  // 1) precision conversion (weights transposed for Bt-fragment loading)
  {
    int n = S * HID;
    cvt_f32_bf16<<<(n + 255) / 256, 256, 0, stream>>>(hidden, Hb, n);
    n = HID * (NH * HD);
    cvt_f32_bf16_T<<<(n + 255) / 256, 256, 0, stream>>>(wq, WqT, HID, NH * HD);
    n = HID * (NKV * HD);
    cvt_f32_bf16_T<<<(n + 255) / 256, 256, 0, stream>>>(wk, WkT, HID, NKV * HD);
    cvt_f32_bf16_T<<<(n + 255) / 256, 256, 0, stream>>>(wv, WvT, HID, NKV * HD);
    n = (NH * HD) * HID;
    cvt_f32_bf16_T<<<(n + 255) / 256, 256, 0, stream>>>(wo, WoT, NH * HD, HID);
  }

  // 2) QKV projections (bf16 WMMA GEMM)
  {
    dim3 gq((NH * HD) / 64, S / 64);
    gemm_wmma<0><<<gq, 128, 0, stream>>>(Hb, WqT, Qb, S, NH * HD, HID, NH * HD);
    dim3 gk((NKV * HD) / 64, S / 64);
    gemm_wmma<0><<<gk, 128, 0, stream>>>(Hb, WkT, Kb, S, NKV * HD, HID, NKV * HD);
    gemm_wmma<1><<<gk, 128, 0, stream>>>(Hb, WvT, Vt, S, NKV * HD, HID, S);
  }

  // 3) RoPE on Q and K (in place)
  {
    int nq = S * NH * (HD / 2);
    rope_kernel<<<(nq + 255) / 256, 256, 0, stream>>>(Qb, posids, NH);
    int nk = S * NKV * (HD / 2);
    rope_kernel<<<(nk + 255) / 256, 256, 0, stream>>>(Kb, posids, NKV);
  }

  // 4) flash attention
  {
    dim3 ga(S / 64, NH);
    attn_wmma<<<ga, 128, 0, stream>>>(Qb, Kb, Vt, AttO);
  }

  // 5) output projection -> fp32 d_out
  {
    dim3 go(HID / 64, S / 64);
    gemm_wmma<2><<<go, 128, 0, stream>>>(AttO, WoT, out, S, HID, NH * HD, HID);
  }
  (void)in_sizes; (void)n_in; (void)out_size;
}
